// FeedForwardQAttention_33621003993529
// MI455X (gfx1250) — matrix operand order repaired and checked by reading, not compiled, not run on hardware
//
#include <hip/hip_runtime.h>
#include <hip/hip_bf16.h>

// Problem constants (B, S, D, H, L, HD)
#define BB 2
#define SS 2048
#define DD 1024
#define HH 16
#define LL 4
#define HDM 64

typedef __attribute__((ext_vector_type(16))) __bf16 v16bf;
typedef __attribute__((ext_vector_type(8)))  float  v8f;
typedef __attribute__((ext_vector_type(4)))  int    v4i;

#define AS1 __attribute__((address_space(1)))
#define AS3 __attribute__((address_space(3)))

union FragB {
    uint4 q[2];
    v16bf v;
    unsigned short us[16];
};

__device__ __forceinline__ unsigned short f2bf(float f) {
    union { float f; unsigned u; } a; a.f = f;
    unsigned u = a.u;
    unsigned r = u + 0x7FFFu + ((u >> 16) & 1u);
    return (unsigned short)(r >> 16);
}

// ---- CDNA5 async global->LDS (ASYNCcnt-tracked), builtin if present else asm
__device__ __forceinline__ void async_copy16(const void* g, const void* l) {
#if __has_builtin(__builtin_amdgcn_global_load_async_to_lds_b128)
    __builtin_amdgcn_global_load_async_to_lds_b128(
        (AS1 v4i*)(unsigned long long)g,
        (AS3 v4i*)(unsigned)(unsigned long long)l, 0, 0);
#else
    unsigned lds32 = (unsigned)(unsigned long long)l;
    asm volatile("global_load_async_to_lds_b128 %0, %1, off"
                 :: "v"(lds32), "v"((unsigned long long)g) : "memory");
#endif
}

__device__ __forceinline__ void wait_async8() {
#if __has_builtin(__builtin_amdgcn_s_wait_asynccnt)
    __builtin_amdgcn_s_wait_asynccnt(8);
#else
    asm volatile("s_wait_asynccnt 0x8" ::: "memory");
#endif
}
__device__ __forceinline__ void wait_async0() {
#if __has_builtin(__builtin_amdgcn_s_wait_asynccnt)
    __builtin_amdgcn_s_wait_asynccnt(0);
#else
    asm volatile("s_wait_asynccnt 0x0" ::: "memory");
#endif
}

// ---------------------------------------------------------------------------
// Kernel 0: f32 -> bf16 elementwise convert (x -> xb), 8 elems/thread
// ---------------------------------------------------------------------------
__global__ __launch_bounds__(256) void cvt_bf16_kernel(const float* __restrict__ src,
                                                       unsigned short* __restrict__ dst) {
    long i = ((long)blockIdx.x * 256 + threadIdx.x) * 8;
    float4 a = *(const float4*)(src + i);
    float4 b = *(const float4*)(src + i + 4);
    union { unsigned short us[8]; uint4 q; } o;
    o.us[0] = f2bf(a.x); o.us[1] = f2bf(a.y); o.us[2] = f2bf(a.z); o.us[3] = f2bf(a.w);
    o.us[4] = f2bf(b.x); o.us[5] = f2bf(b.y); o.us[6] = f2bf(b.z); o.us[7] = f2bf(b.w);
    *(uint4*)(dst + i) = o.q;
}

// ---------------------------------------------------------------------------
// Kernel 1/3: C = act(A @ W^T + bias)
//   A: Mtot x K (f32, row-major), W: N x K (f32, row-major)
//   MODE 0: store f32 (Cf) and bf16 (Cb) copies, no activation  (Q projection)
//   MODE 1: store f32 with exact GELU                           (output proj)
// Block: 256 thr (8 waves), tile 128x128, K-step 32, bf16 WMMA.
// Double-buffered f32 LDS tiles filled by async global->LDS DMA.
// ---------------------------------------------------------------------------
template<int MODE>
__global__ __launch_bounds__(256) void gemm_nt_bf16_kernel(
    const float* __restrict__ A, const float* __restrict__ W,
    const float* __restrict__ bias,
    float* __restrict__ Cf, unsigned short* __restrict__ Cb,
    int Mtot, int N, int K)
{
    __shared__ float AbufF[2][128 * 32];
    __shared__ float WbufF[2][128 * 32];

    const int t    = threadIdx.x;
    const int lane = t & 31;
    const int w    = t >> 5;
    const int wr   = w >> 2;       // 0..1
    const int wc   = w & 3;        // 0..3
    const int hi   = lane >> 4;    // 0..1
    const int l15  = lane & 15;
    const long blockM = (long)blockIdx.x * 128;
    const long blockN = (long)blockIdx.y * 128;

    v8f acc[4][2];
    #pragma unroll
    for (int mt = 0; mt < 4; ++mt)
        #pragma unroll
        for (int nt = 0; nt < 2; ++nt) { v8f z = {}; acc[mt][nt] = z; }

    const int steps = K / 32;

    // Issue one stage: each thread queues 4 b128 async copies for A + 4 for W
    auto stage = [&](int it, int buf) {
        const int kk = it * 32;
        #pragma unroll
        for (int i = 0; i < 4; ++i) {
            int e  = t + 256 * i;          // float4 slot 0..1023
            int r  = e >> 3;               // row 0..127
            int c4 = (e & 7) * 4;          // col 0..28
            async_copy16(&A[(blockM + r) * (long)K + kk + c4], &AbufF[buf][e * 4]);
        }
        #pragma unroll
        for (int i = 0; i < 4; ++i) {
            int e  = t + 256 * i;
            int r  = e >> 3;
            int c4 = (e & 7) * 4;
            async_copy16(&W[(blockN + r) * (long)K + kk + c4], &WbufF[buf][e * 4]);
        }
    };

    stage(0, 0);

    for (int it = 0; it < steps; ++it) {
        const int p = it & 1;
        const bool more = (it + 1 < steps);
        if (more) stage(it + 1, p ^ 1);
        if (more) wait_async8(); else wait_async0();
        __syncthreads();   // tile p resident in LDS for all waves

        // A fragments (16x32 M x K): halfs j -> K = 16*hi + j (contiguous)
        FragB aF[4];
        #pragma unroll
        for (int mt = 0; mt < 4; ++mt) {
            int row = wr * 64 + mt * 16 + l15;
            int kb  = hi * 16;
            const float* bse = &AbufF[p][row * 32];
            float4 x0 = *(const float4*)&bse[kb];
            float4 x1 = *(const float4*)&bse[kb + 4];
            float4 x2 = *(const float4*)&bse[kb + 8];
            float4 x3 = *(const float4*)&bse[kb + 12];
            aF[mt].us[0]  = f2bf(x0.x); aF[mt].us[1]  = f2bf(x0.y);
            aF[mt].us[2]  = f2bf(x0.z); aF[mt].us[3]  = f2bf(x0.w);
            aF[mt].us[4]  = f2bf(x1.x); aF[mt].us[5]  = f2bf(x1.y);
            aF[mt].us[6]  = f2bf(x1.z); aF[mt].us[7]  = f2bf(x1.w);
            aF[mt].us[8]  = f2bf(x2.x); aF[mt].us[9]  = f2bf(x2.y);
            aF[mt].us[10] = f2bf(x2.z); aF[mt].us[11] = f2bf(x2.w);
            aF[mt].us[12] = f2bf(x3.x); aF[mt].us[13] = f2bf(x3.y);
            aF[mt].us[14] = f2bf(x3.z); aF[mt].us[15] = f2bf(x3.w);
        }
        // B fragments (32x16 K x N): halfs j -> K = 16*hi + j (contiguous)
        FragB bF[2];
        #pragma unroll
        for (int nt = 0; nt < 2; ++nt) {
            int row = wc * 32 + nt * 16 + l15;
            int kb  = hi * 16;
            const float* bse = &WbufF[p][row * 32 + kb];
            #pragma unroll
            for (int j4 = 0; j4 < 4; ++j4) {
                float4 y = *(const float4*)&bse[j4 * 4];
                bF[nt].us[j4 * 4 + 0] = f2bf(y.x);
                bF[nt].us[j4 * 4 + 1] = f2bf(y.y);
                bF[nt].us[j4 * 4 + 2] = f2bf(y.z);
                bF[nt].us[j4 * 4 + 3] = f2bf(y.w);
            }
        }
        #pragma unroll
        for (int mt = 0; mt < 4; ++mt)
            #pragma unroll
            for (int nt = 0; nt < 2; ++nt)
                acc[mt][nt] = __builtin_amdgcn_wmma_f32_16x16x32_bf16(
                    false, aF[mt].v, false, bF[nt].v, (short)0, acc[mt][nt], false, false);
        __syncthreads();   // reads done; buffer p may be overwritten next issue
    }

    // Epilogue: bias (+ GELU or bf16 copy)
    #pragma unroll
    for (int mt = 0; mt < 4; ++mt) {
        #pragma unroll
        for (int nt = 0; nt < 2; ++nt) {
            #pragma unroll
            for (int i = 0; i < 8; ++i) {
                long grow = blockM + wr * 64 + mt * 16 + i + 8 * hi;
                long gcol = blockN + wc * 32 + nt * 16 + l15;
                float v = acc[mt][nt][i] + bias[gcol];
                if (MODE == 0) {
                    Cf[grow * N + gcol] = v;
                    Cb[grow * N + gcol] = f2bf(v);
                } else {
                    float g = 0.5f * v * (1.0f + erff(v * 0.70710678118654752f));
                    Cf[grow * N + gcol] = g;
                }
            }
        }
    }
}

// ---------------------------------------------------------------------------
// Kernel 2: causal self-attention (q@q^T) with 4-slot memory attention,
// flash-style online softmax. One wave (32 thr) per 16-row query tile.
// grid.x = B*H*(S/16). Writes ctx (B,S,D) f32.
// ---------------------------------------------------------------------------
__global__ __launch_bounds__(32) void attn_kernel(
    const unsigned short* __restrict__ qb,   // (B*S, D) bf16
    const float* __restrict__ q32,           // (B*S, D) f32
    const unsigned short* __restrict__ xb,   // (B*S, D) bf16  (V source)
    const float* __restrict__ past_q,        // (B,H,S,L,HD) f32
    const float* __restrict__ past_v,        // (B,H,S,L,HD) f32
    float* __restrict__ ctx)                 // (B*S, D) f32
{
    __shared__ unsigned short Pt[16 * 32];   // P tile, bf16, 16 rows x 32 keys
    __shared__ unsigned short Vt[64 * 32];   // V^T tile, bf16: [d][k]
    __shared__ float Ct[16 * 64];            // context tile f32
    __shared__ float RMax[16];
    __shared__ float RSum[16];

    const int lane = threadIdx.x;
    const int l15  = lane & 15;
    const int hi   = lane >> 4;

    const int qt = blockIdx.x & 127;
    const int h  = (blockIdx.x >> 7) & 15;
    const int b  = blockIdx.x >> 11;
    const int m0 = qt * 16;

    const float inv = 0.125f;  // 1/sqrt(HD)

    // A fragments of Q (16 rows x HD=64 -> two 16x32 frags)
    FragB aQ0, aQ1;
    {
        const unsigned short* qrow = qb + ((long)(b * SS + m0 + l15)) * DD + h * HDM;
        int kb = hi * 16;
        aQ0.q[0] = *(const uint4*)&qrow[kb];
        aQ0.q[1] = *(const uint4*)&qrow[kb + 8];
        aQ1.q[0] = *(const uint4*)&qrow[32 + kb];
        aQ1.q[1] = *(const uint4*)&qrow[32 + kb + 8];
    }

    v8f c0 = {}, c1 = {}, c2 = {}, c3 = {};
    float rmax[8], rsum[8];
    #pragma unroll
    for (int i = 0; i < 8; ++i) { rmax[i] = -3.0e38f; rsum[i] = 0.0f; }

    const int nchunk = ((m0 + 15) >> 5) + 1;   // 32-key chunks with kbase <= m0+15

    for (int cch = 0; cch < nchunk; ++cch) {
        const int k0 = cch * 32;
        __syncthreads();  // protect Vt/Pt reuse (single-wave: s_barrier==NOP + fence)

        // Prefetch next chunk's K/V rows
        if (cch + 1 < nchunk) {
            __builtin_prefetch(qb + ((long)(b * SS + k0 + 32 + l15)) * DD + h * HDM, 0, 0);
            __builtin_prefetch(xb + ((long)(b * SS + k0 + 32 + lane)) * DD + h * HDM, 0, 0);
        }

        // Stage V chunk transposed into LDS: lane loads key row (64 halfs), scatters
        {
            const unsigned short* vrow = xb + ((long)(b * SS + k0 + lane)) * DD + h * HDM;
            union { uint4 q[8]; unsigned short us[64]; } vv;
            #pragma unroll
            for (int j = 0; j < 8; ++j) vv.q[j] = ((const uint4*)vrow)[j];
            #pragma unroll
            for (int d = 0; d < 64; ++d) Vt[d * 32 + lane] = vv.us[d];
        }

        // Score tiles: S0 for keys k0..k0+15, S1 for k0+16..k0+31
        FragB bK0a, bK0b, bK1a, bK1b;
        {
            const unsigned short* kr0 = qb + ((long)(b * SS + k0 + l15)) * DD + h * HDM;
            const unsigned short* kr1 = kr0 + 16 * DD;
            int kb = hi * 16;
            bK0a.q[0] = *(const uint4*)&kr0[kb];      bK0a.q[1] = *(const uint4*)&kr0[kb + 8];
            bK0b.q[0] = *(const uint4*)&kr0[32 + kb]; bK0b.q[1] = *(const uint4*)&kr0[32 + kb + 8];
            bK1a.q[0] = *(const uint4*)&kr1[kb];      bK1a.q[1] = *(const uint4*)&kr1[kb + 8];
            bK1b.q[0] = *(const uint4*)&kr1[32 + kb]; bK1b.q[1] = *(const uint4*)&kr1[32 + kb + 8];
        }
        v8f s0 = {}, s1 = {};
        s0 = __builtin_amdgcn_wmma_f32_16x16x32_bf16(false, aQ0.v, false, bK0a.v, (short)0, s0, false, false);
        s0 = __builtin_amdgcn_wmma_f32_16x16x32_bf16(false, aQ1.v, false, bK0b.v, (short)0, s0, false, false);
        s1 = __builtin_amdgcn_wmma_f32_16x16x32_bf16(false, aQ0.v, false, bK1a.v, (short)0, s1, false, false);
        s1 = __builtin_amdgcn_wmma_f32_16x16x32_bf16(false, aQ1.v, false, bK1b.v, (short)0, s1, false, false);

        // Causal mask + scale, per-chunk row max
        float cmax[8];
        #pragma unroll
        for (int i = 0; i < 8; ++i) {
            int row  = m0 + i + 8 * hi;
            int key0 = k0 + l15;
            int key1 = k0 + 16 + l15;
            float v0 = (key0 <= row) ? s0[i] * inv : -3.0e38f;
            float v1 = (key1 <= row) ? s1[i] * inv : -3.0e38f;
            s0[i] = v0; s1[i] = v1;
            cmax[i] = fmaxf(v0, v1);
        }
        #pragma unroll
        for (int off = 1; off < 16; off <<= 1)
            #pragma unroll
            for (int i = 0; i < 8; ++i)
                cmax[i] = fmaxf(cmax[i], __shfl_xor(cmax[i], off, 32));

        // Online softmax update + rescale context accumulators
        float psum[8];
        #pragma unroll
        for (int i = 0; i < 8; ++i) {
            float nm = fmaxf(rmax[i], cmax[i]);
            float sc = __expf(rmax[i] - nm);
            rmax[i] = nm;
            rsum[i] *= sc;
            c0[i] *= sc; c1[i] *= sc; c2[i] *= sc; c3[i] *= sc;
            float p0 = __expf(s0[i] - nm);
            float p1 = __expf(s1[i] - nm);
            s0[i] = p0; s1[i] = p1;
            psum[i] = p0 + p1;
        }
        #pragma unroll
        for (int off = 1; off < 16; off <<= 1)
            #pragma unroll
            for (int i = 0; i < 8; ++i)
                psum[i] += __shfl_xor(psum[i], off, 32);
        #pragma unroll
        for (int i = 0; i < 8; ++i) rsum[i] += psum[i];

        // Store P (bf16) to LDS in row-major 16x32 for A-fragment reload
        #pragma unroll
        for (int i = 0; i < 8; ++i) {
            Pt[(i + 8 * hi) * 32 + l15]      = f2bf(s0[i]);
            Pt[(i + 8 * hi) * 32 + 16 + l15] = f2bf(s1[i]);
        }
        __syncthreads();

        // A fragment of P, B fragments of V^T, accumulate context
        FragB aP;
        {
            int kb = hi * 16;
            aP.q[0] = *(const uint4*)&Pt[l15 * 32 + kb];
            aP.q[1] = *(const uint4*)&Pt[l15 * 32 + kb + 8];
        }
        #pragma unroll
        for (int n = 0; n < 4; ++n) {
            FragB bV;
            int d  = n * 16 + l15;
            int kb = hi * 16;
            bV.q[0] = *(const uint4*)&Vt[d * 32 + kb];
            bV.q[1] = *(const uint4*)&Vt[d * 32 + kb + 8];
            v8f* cc = (n == 0) ? &c0 : (n == 1) ? &c1 : (n == 2) ? &c2 : &c3;
            *cc = __builtin_amdgcn_wmma_f32_16x16x32_bf16(false, aP.v, false, bV.v, (short)0, *cc, false, false);
        }
    }

    // Spill context tile + softmax state to LDS
    #pragma unroll
    for (int i = 0; i < 8; ++i) {
        Ct[(i + 8 * hi) * 64 +  0 + l15] = c0[i];
        Ct[(i + 8 * hi) * 64 + 16 + l15] = c1[i];
        Ct[(i + 8 * hi) * 64 + 32 + l15] = c2[i];
        Ct[(i + 8 * hi) * 64 + 48 + l15] = c3[i];
    }
    if (l15 == 0) {
        #pragma unroll
        for (int i = 0; i < 8; ++i) { RMax[8 * hi + i] = rmax[i]; RSum[8 * hi + i] = rsum[i]; }
    }
    __syncthreads();

    // Memory-slot attention + final normalize: 2 lanes per row, 32 d-cols each
    {
        const int r    = lane >> 1;
        const int half = lane & 1;
        const int s    = m0 + r;
        const long pqb = ((((long)b * HH + h) * SS + s) * LL) * HDM;
        const float4* qr = (const float4*)(q32 + ((long)b * SS + s) * DD + h * HDM);

        float ms[2];
        #pragma unroll
        for (int t2 = 0; t2 < 2; ++t2) {
            int l = half * 2 + t2;
            const float4* pq = (const float4*)(past_q + pqb + (long)l * HDM);
            float acc = 0.0f;
            #pragma unroll
            for (int j = 0; j < 16; ++j) {
                float4 a = qr[j], p = pq[j];
                acc += a.x * p.x + a.y * p.y + a.z * p.z + a.w * p.w;
            }
            ms[t2] = acc * inv;
        }
        float mymax = fmaxf(ms[0], ms[1]);
        float omax  = __shfl_xor(mymax, 1, 32);
        float rm = RMax[r];
        float rs = RSum[r];
        float fm = fmaxf(fmaxf(mymax, omax), rm);
        float sc = __expf(rm - fm);
        float p0 = __expf(ms[0] - fm);
        float p1 = __expf(ms[1] - fm);
        float pin = p0 + p1;
        float tot = rs * sc + pin + __shfl_xor(pin, 1, 32);
        float op0 = __shfl_xor(p0, 1, 32);
        float op1 = __shfl_xor(p1, 1, 32);
        float pl[4];
        if (half == 0) { pl[0] = p0; pl[1] = p1; pl[2] = op0; pl[3] = op1; }
        else           { pl[0] = op0; pl[1] = op1; pl[2] = p0; pl[3] = p1; }
        float invtot = 1.0f / tot;

        const int d0 = half * 32;
        const float* pv = past_v + pqb;
        float* outrow = ctx + ((long)b * SS + s) * DD + h * HDM + d0;
        #pragma unroll
        for (int j = 0; j < 8; ++j) {
            float4 cv = *(const float4*)&Ct[r * 64 + d0 + 4 * j];
            float4 o;
            o.x = cv.x * sc; o.y = cv.y * sc; o.z = cv.z * sc; o.w = cv.w * sc;
            #pragma unroll
            for (int l = 0; l < 4; ++l) {
                float4 pvv = *(const float4*)&pv[l * HDM + d0 + 4 * j];
                o.x += pl[l] * pvv.x; o.y += pl[l] * pvv.y;
                o.z += pl[l] * pvv.z; o.w += pl[l] * pvv.w;
            }
            o.x *= invtot; o.y *= invtot; o.z *= invtot; o.w *= invtot;
            *(float4*)&outrow[4 * j] = o;
        }
    }
}

// ---------------------------------------------------------------------------
extern "C" void kernel_launch(void* const* d_in, const int* in_sizes, int n_in,
                              void* d_out, int out_size, void* d_ws, size_t ws_size,
                              hipStream_t stream) {
    const float* x      = (const float*)d_in[0];
    const float* past_q = (const float*)d_in[1];
    const float* past_v = (const float*)d_in[2];
    const float* Wq     = (const float*)d_in[3];
    const float* bq     = (const float*)d_in[4];
    const float* Wo     = (const float*)d_in[5];
    const float* bo     = (const float*)d_in[6];
    float* out = (float*)d_out;

    char* wsb = (char*)d_ws;
    const long NTOK = (long)BB * SS;          // 4096
    const long NEL  = NTOK * DD;              // 4 Mi elements
    float*          q32  = (float*)(wsb + 0);
    float*          ctxb = (float*)(wsb + NEL * 4);
    unsigned short* qb   = (unsigned short*)(wsb + NEL * 8);
    unsigned short* xb   = (unsigned short*)(wsb + NEL * 8 + NEL * 2);

    // 0) x -> bf16
    cvt_bf16_kernel<<<(int)(NEL / (256 * 8)), 256, 0, stream>>>(x, xb);

    // 1) Q = x @ Wq^T + bq   (f32 + bf16 copies)
    gemm_nt_bf16_kernel<0><<<dim3(NTOK / 128, DD / 128), 256, 0, stream>>>(
        x, Wq, bq, q32, qb, (int)NTOK, DD, DD);

    // 2) attention -> ctx
    attn_kernel<<<BB * HH * (SS / 16), 32, 0, stream>>>(qb, q32, xb, past_q, past_v, ctxb);

    // 3) out = gelu(ctx @ Wo^T + bo)
    gemm_nt_bf16_kernel<1><<<dim3(NTOK / 128, DD / 128), 256, 0, stream>>>(
        ctxb, Wo, bo, out, nullptr, (int)NTOK, DD, DD);
}
